// CausalSelfAttention_67095979098307
// MI455X (gfx1250) — compile-verified, matrix operand order checked by reading, not run on hardware
//
#include <hip/hip_runtime.h>
#include <hip/hip_bf16.h>

#define BATCH 4
#define TSEQ  2048
#define CDIM  576
#define HQ    9
#define HKVN  3
#define DHEAD 64
#define NQKV  960   // 576 q + 192 k + 192 v

typedef __attribute__((ext_vector_type(16))) _Float16 v16h;
typedef __attribute__((ext_vector_type(8)))  _Float16 v8h;
typedef __attribute__((ext_vector_type(8)))  float    v8f;

__device__ __forceinline__ v8f wmma16(v16h a, v16h b, v8f c) {
  return __builtin_amdgcn_wmma_f32_16x16x32_f16(false, a, false, b, (short)0, c,
                                                false, false);
}

__device__ __forceinline__ v8f zero8() {
  v8f z;
#pragma unroll
  for (int i = 0; i < 8; ++i) z[i] = 0.0f;
  return z;
}

// Load one 16x32 A/B fragment from a row-major f16 matrix (ISA 7.12.2 layout).
__device__ __forceinline__ v16h load_ab(const _Float16* rowk, int lane) {
  const _Float16* p = rowk + ((lane >> 4) << 3);
  v8h lo = *(const v8h*)(p);
  v8h hi = *(const v8h*)(p + 16);
  v16h f;
#pragma unroll
  for (int i = 0; i < 8; ++i) { f[i] = lo[i]; f[8 + i] = hi[i]; }
  return f;
}

// CDNA5 async DMA: global -> LDS, tracked by ASYNCcnt (no VGPR round trip).
__device__ __forceinline__ void async_b128(_Float16* ldst,
                                           const _Float16* gsrc) {
  unsigned l = (unsigned)(uintptr_t)(void*)ldst;  // low 32 bits = LDS offset
  asm volatile("global_load_async_to_lds_b128 %0, %1, off"
               :: "v"(l), "v"(gsrc) : "memory");
}
__device__ __forceinline__ void wait_async0() {
  asm volatile("s_wait_asynccnt 0" ::: "memory");
}

// ---------------- conversion / packing / rope table ----------------

__global__ void f32_to_f16_kernel(const float* __restrict__ s,
                                  _Float16* __restrict__ d, int n) {
  int i = blockIdx.x * blockDim.x + threadIdx.x;
  if (i < n) d[i] = (_Float16)s[i];
}

__global__ void pack_wqkv_kernel(const float* __restrict__ wq,
                                 const float* __restrict__ wk,
                                 const float* __restrict__ wv,
                                 _Float16* __restrict__ d) {
  int i = blockIdx.x * blockDim.x + threadIdx.x;
  if (i >= NQKV * CDIM) return;
  int r = i / CDIM, c = i - r * CDIM;
  float v;
  if (r < CDIM)       v = wq[(size_t)r * CDIM + c];
  else if (r < 768)   v = wk[(size_t)(r - 576) * CDIM + c];
  else                v = wv[(size_t)(r - 768) * CDIM + c];
  d[i] = (_Float16)v;
}

__global__ void rope_table_kernel(float* __restrict__ ct,
                                  float* __restrict__ st) {
  int i = blockIdx.x * blockDim.x + threadIdx.x;  // t*32 + f
  if (i >= TSEQ * 32) return;
  int t = i >> 5, f = i & 31;
  float fr  = __expf(-0.28782313662425572f * (float)f);  // theta^(-f/32)
  float ang = (float)t * fr;
  ct[i] = cosf(ang);
  st[i] = sinf(ang);
}

// ---------------- fused QKV GEMM + RoPE epilogue ----------------
// B (weights) staged via double-buffered async-to-LDS, shared by all 8 waves.
__global__ __launch_bounds__(256) void qkv_gemm_rope_kernel(
    const _Float16* __restrict__ xh, const _Float16* __restrict__ w,
    const float* __restrict__ ct, const float* __restrict__ st,
    _Float16* __restrict__ qh, _Float16* __restrict__ kh,
    _Float16* __restrict__ vt) {
  __shared__ __align__(16) _Float16 sB[2][64 * 32];  // 64 n-rows x 32 k
  const int lane = threadIdx.x & 31;
  const int wv   = threadIdx.x >> 5;
  const int m0   = blockIdx.x * 128 + wv * 16;
  const int by   = blockIdx.y;
  const int n0   = by * 64;

  v8f acc[4];
#pragma unroll
  for (int j = 0; j < 4; ++j) acc[j] = zero8();

  const int brow = threadIdx.x >> 2;         // 0..63
  const int bcol = (threadIdx.x & 3) << 3;   // 0,8,16,24 halves
  const _Float16* bsrc = w + (size_t)(n0 + brow) * CDIM + bcol;
  async_b128(&sB[0][brow * 32 + bcol], bsrc);

  const _Float16* abase = xh + (size_t)(m0 + (lane & 15)) * CDIM;
  int cur = 0;
  for (int k0 = 0; k0 < CDIM; k0 += 32) {
    wait_async0();
    __syncthreads();
    if (k0 + 32 < CDIM)
      async_b128(&sB[cur ^ 1][brow * 32 + bcol], bsrc + (k0 + 32));
    v16h a = load_ab(abase + k0, lane);
#pragma unroll
    for (int j = 0; j < 4; ++j) {
      v16h b = load_ab(&sB[cur][(j * 16 + (lane & 15)) * 32], lane);
      acc[j] = wmma16(a, b, acc[j]);
    }
    cur ^= 1;
  }

  // epilogue: RoPE (q,k) from table + scatter; v stored transposed [d][t]
  const int mhi = (lane >> 4) << 3;
  if (by < 12) {  // Q or K head: RoPE
#pragma unroll
    for (int j = 0; j < 4; ++j) {
#pragma unroll
      for (int r = 0; r < 8; ++r) {
        float vval = acc[j][r];
        float pv   = __shfl_xor(vval, 1);   // RoPE partner (adjacent n lane)
        int n  = lane & 15;
        int d  = j * 16 + n;
        int mg = m0 + r + mhi;
        int bi = mg >> 11;
        int t  = mg & 2047;
        int ti = t * 32 + (d >> 1);
        float cs = ct[ti], sn = st[ti];
        float o = (n & 1) ? (pv * sn + vval * cs) : (vval * cs - pv * sn);
        if (by < HQ) {
          o *= 0.125f;                      // fold 1/sqrt(D)
          qh[(((size_t)bi * HQ + by) * TSEQ + t) * DHEAD + d] = (_Float16)o;
        } else {
          kh[(((size_t)bi * HKVN + (by - HQ)) * TSEQ + t) * DHEAD + d] =
              (_Float16)o;
        }
      }
    }
  } else {        // V head (transposed store)
#pragma unroll
    for (int j = 0; j < 4; ++j) {
#pragma unroll
      for (int r = 0; r < 8; ++r) {
        int d  = j * 16 + (lane & 15);
        int mg = m0 + r + mhi;
        int bi = mg >> 11;
        int t  = mg & 2047;
        vt[(((size_t)bi * HKVN + (by - 12)) * DHEAD + d) * TSEQ + t] =
            (_Float16)acc[j][r];
      }
    }
  }
}

// ---------------- flash attention ----------------
// 8 waves x 16 q-rows; K/V tiles double-buffered via async-to-LDS DMA.
__global__ __launch_bounds__(256) void attn_kernel(
    const _Float16* __restrict__ qh, const _Float16* __restrict__ kh,
    const _Float16* __restrict__ vt, _Float16* __restrict__ yh) {
  __shared__ __align__(16) _Float16 sK[2][64 * 64];
  __shared__ __align__(16) _Float16 sV[2][64 * 64];
  __shared__ __align__(16) _Float16 sP[8 * 16 * 64];

  const int lane = threadIdx.x & 31;
  const int wv   = threadIdx.x >> 5;
  const int bid  = blockIdx.x;
  const int qb   = bid & 15;
  const int hh   = (bid >> 4) % HQ;
  const int b    = bid / (16 * HQ);
  const int hk   = hh / 3;
  const int tq0  = qb * 128 + wv * 16;

  const _Float16* qbase =
      qh + (((size_t)b * HQ + hh) * TSEQ + tq0 + (lane & 15)) * DHEAD;
  v16h aq0 = load_ab(qbase, lane);
  v16h aq1 = load_ab(qbase + 32, lane);

  const _Float16* kbase = kh + ((size_t)b * HKVN + hk) * TSEQ * DHEAD;
  const _Float16* vbase = vt + ((size_t)b * HKVN + hk) * (size_t)DHEAD * TSEQ;

  v8f oacc[4];
#pragma unroll
  for (int j = 0; j < 4; ++j) oacc[j] = zero8();
  float mrow[8], lrow[8];
#pragma unroll
  for (int r = 0; r < 8; ++r) { mrow[r] = -3.0e38f; lrow[r] = 0.0f; }

  const int ldr = threadIdx.x >> 2;          // 0..63
  const int ldc = (threadIdx.x & 3) << 4;    // 0,16,32,48 halves
  const int nkt = qb * 2 + 2;

  auto issueKV = [&](int kt0, int buf) {
    const _Float16* sk = kbase + (size_t)(kt0 + ldr) * DHEAD + ldc;
    async_b128(&sK[buf][ldr * 64 + ldc],     sk);
    async_b128(&sK[buf][ldr * 64 + ldc + 8], sk + 8);
    const _Float16* sv = vbase + (size_t)ldr * TSEQ + kt0 + ldc;
    async_b128(&sV[buf][ldr * 64 + ldc],     sv);
    async_b128(&sV[buf][ldr * 64 + ldc + 8], sv + 8);
  };
  issueKV(0, 0);
  int cur = 0;

  for (int kt = 0; kt < nkt; ++kt) {
    const int kt0 = kt * 64;
    wait_async0();
    __syncthreads();
    if (kt + 1 < nkt) issueKV(kt0 + 64, cur ^ 1);  // overlap DMA with compute

    if (kt0 <= tq0 + 15) {
      // S = q * k^T  (scale folded into q)
      v8f sacc[4];
#pragma unroll
      for (int j = 0; j < 4; ++j) {
        const _Float16* kb = &sK[cur][(j * 16 + (lane & 15)) * 64];
        v16h b0 = load_ab(kb, lane);
        v16h b1 = load_ab(kb + 32, lane);
        v8f s = zero8();
        s = wmma16(aq0, b0, s);
        s = wmma16(aq1, b1, s);
        sacc[j] = s;
      }
      if (kt0 + 63 > tq0) {  // causal mask
#pragma unroll
        for (int j = 0; j < 4; ++j)
#pragma unroll
          for (int r = 0; r < 8; ++r) {
            int kv = kt0 + j * 16 + (lane & 15);
            int tq = tq0 + r + ((lane >> 4) << 3);
            if (kv > tq) sacc[j][r] = -3.0e38f;
          }
      }
      // online softmax (rows live in 16-lane groups)
      float tmax[8];
#pragma unroll
      for (int r = 0; r < 8; ++r)
        tmax[r] = fmaxf(fmaxf(sacc[0][r], sacc[1][r]),
                        fmaxf(sacc[2][r], sacc[3][r]));
#pragma unroll
      for (int off = 8; off >= 1; off >>= 1)
#pragma unroll
        for (int r = 0; r < 8; ++r)
          tmax[r] = fmaxf(tmax[r], __shfl_xor(tmax[r], off));

      float psum[8];
      _Float16* prow = &sP[wv * 1024];
#pragma unroll
      for (int r = 0; r < 8; ++r) {
        float mn = fmaxf(mrow[r], tmax[r]);
        float sc = __expf(mrow[r] - mn);
        mrow[r] = mn;
        lrow[r] *= sc;
#pragma unroll
        for (int dj = 0; dj < 4; ++dj) oacc[dj][r] *= sc;
        float ps = 0.0f;
        int rr = (r + ((lane >> 4) << 3)) * 64 + (lane & 15);
#pragma unroll
        for (int j = 0; j < 4; ++j) {
          float p = __expf(sacc[j][r] - mn);
          ps += p;
          prow[rr + j * 16] = (_Float16)p;   // stage P for A-layout reload
        }
        psum[r] = ps;
      }
#pragma unroll
      for (int off = 8; off >= 1; off >>= 1)
#pragma unroll
        for (int r = 0; r < 8; ++r) psum[r] += __shfl_xor(psum[r], off);
#pragma unroll
      for (int r = 0; r < 8; ++r) lrow[r] += psum[r];

      // O += P * V (P reloaded in A-fragment layout from wave-private LDS)
      const _Float16* pb = &sP[wv * 1024 + (lane & 15) * 64];
      v16h ap0 = load_ab(pb, lane);
      v16h ap1 = load_ab(pb + 32, lane);
#pragma unroll
      for (int dj = 0; dj < 4; ++dj) {
        const _Float16* vb = &sV[cur][(dj * 16 + (lane & 15)) * 64];
        v16h b0 = load_ab(vb, lane);
        v16h b1 = load_ab(vb + 32, lane);
        oacc[dj] = wmma16(ap0, b0, oacc[dj]);
        oacc[dj] = wmma16(ap1, b1, oacc[dj]);
      }
    }
    cur ^= 1;
  }

  // normalize and store y [B, T, C]
#pragma unroll
  for (int r = 0; r < 8; ++r) {
    int t = tq0 + r + ((lane >> 4) << 3);
    float inv = 1.0f / lrow[r];
    _Float16* dst = yh + ((size_t)b * TSEQ + t) * CDIM + hh * DHEAD;
#pragma unroll
    for (int dj = 0; dj < 4; ++dj)
      dst[dj * 16 + (lane & 15)] = (_Float16)(oacc[dj][r] * inv);
  }
}

// ---------------- output projection (fp32 out) ----------------
__global__ __launch_bounds__(256) void out_proj_kernel(
    const _Float16* __restrict__ yh, const _Float16* __restrict__ wo,
    float* __restrict__ out) {
  __shared__ __align__(16) _Float16 sB[2][64 * 32];
  const int lane = threadIdx.x & 31;
  const int wv   = threadIdx.x >> 5;
  const int m0   = blockIdx.x * 128 + wv * 16;
  const int n0   = blockIdx.y * 64;

  v8f acc[4];
#pragma unroll
  for (int j = 0; j < 4; ++j) acc[j] = zero8();

  const int brow = threadIdx.x >> 2;
  const int bcol = (threadIdx.x & 3) << 3;
  const _Float16* bsrc = wo + (size_t)(n0 + brow) * CDIM + bcol;
  async_b128(&sB[0][brow * 32 + bcol], bsrc);

  const _Float16* abase = yh + (size_t)(m0 + (lane & 15)) * CDIM;
  int cur = 0;
  for (int k0 = 0; k0 < CDIM; k0 += 32) {
    wait_async0();
    __syncthreads();
    if (k0 + 32 < CDIM)
      async_b128(&sB[cur ^ 1][brow * 32 + bcol], bsrc + (k0 + 32));
    v16h a = load_ab(abase + k0, lane);
#pragma unroll
    for (int j = 0; j < 4; ++j) {
      v16h b = load_ab(&sB[cur][(j * 16 + (lane & 15)) * 32], lane);
      acc[j] = wmma16(a, b, acc[j]);
    }
    cur ^= 1;
  }
#pragma unroll
  for (int j = 0; j < 4; ++j)
#pragma unroll
    for (int r = 0; r < 8; ++r) {
      int mg = m0 + r + ((lane >> 4) << 3);
      out[(size_t)mg * CDIM + n0 + j * 16 + (lane & 15)] = acc[j][r];
    }
}

// ---------------- host ----------------

extern "C" void kernel_launch(void* const* d_in, const int* in_sizes, int n_in,
                              void* d_out, int out_size, void* d_ws,
                              size_t ws_size, hipStream_t stream) {
  const float* x  = (const float*)d_in[0];
  const float* Wq = (const float*)d_in[1];
  const float* Wk = (const float*)d_in[2];
  const float* Wv = (const float*)d_in[3];
  const float* Wo = (const float*)d_in[4];

  char* ws = (char*)d_ws;
  _Float16* xh   = (_Float16*)(ws);                 //  9,437,184
  _Float16* wqkv = (_Float16*)(ws + 9437184);       //  1,105,920
  _Float16* woh  = (_Float16*)(ws + 10543104);      //    663,552
  _Float16* qh   = (_Float16*)(ws + 11206656);      //  9,437,184
  _Float16* kh   = (_Float16*)(ws + 20643840);      //  3,145,728
  _Float16* vt   = (_Float16*)(ws + 23789568);      //  3,145,728
  float*    ct   = (float*)(ws + 26935296);         //    262,144
  float*    st   = (float*)(ws + 27197440);         //    262,144
  _Float16* yh   = xh;  // alias: x no longer needed after the QKV GEMM

  const int nx  = BATCH * TSEQ * CDIM;
  const int nwo = CDIM * CDIM;
  f32_to_f16_kernel<<<(nx + 255) / 256, 256, 0, stream>>>(x, xh, nx);
  pack_wqkv_kernel<<<(NQKV * CDIM + 255) / 256, 256, 0, stream>>>(Wq, Wk, Wv,
                                                                  wqkv);
  f32_to_f16_kernel<<<(nwo + 255) / 256, 256, 0, stream>>>(Wo, woh, nwo);
  rope_table_kernel<<<(TSEQ * 32 + 255) / 256, 256, 0, stream>>>(ct, st);

  dim3 g1(BATCH * TSEQ / 128, NQKV / 64);   // (64, 15)
  qkv_gemm_rope_kernel<<<g1, 256, 0, stream>>>(xh, wqkv, ct, st, qh, kh, vt);

  attn_kernel<<<BATCH * HQ * (TSEQ / 128), 256, 0, stream>>>(qh, kh, vt, yh);

  dim3 g2(BATCH * TSEQ / 128, CDIM / 64);   // (64, 9)
  out_proj_kernel<<<g2, 256, 0, stream>>>(yh, woh, (float*)d_out);
}
